// VectorQuantizer_10788957847717
// MI455X (gfx1250) — matrix operand order repaired and checked by reading, not compiled, not run on hardware
//
#include <hip/hip_runtime.h>

// ---------------- problem constants ----------------
#define VQ_D      512
#define VQ_NROWS  16384      // 16 * 1024
#define VQ_K      8192
#define VQ_CHUNKS 16         // 512 / 32 k-chunks per WMMA pass
#define VQ_SEGS   4          // codebook split along K for grid parallelism
#define VQ_KSEG   (VQ_K / VQ_SEGS)          // 2048 codes per segment
#define VQ_TILES  (VQ_KSEG / 16)            // 128 code tiles per segment

typedef __attribute__((ext_vector_type(16))) __bf16 v16bf;
typedef __attribute__((ext_vector_type(8)))  __bf16 v8bf;
typedef __attribute__((ext_vector_type(8)))  float  v8f;
typedef __attribute__((ext_vector_type(4)))  int    v4i;

// CDNA5 async global->LDS staging (ASYNCcnt-tracked), with safe fallback.
#if defined(__has_builtin)
#if __has_builtin(__builtin_amdgcn_global_load_async_to_lds_b128)
#define VQ_ASYNC 1
#endif
#endif
#ifndef VQ_ASYNC
#define VQ_ASYNC 0
#endif

// builtin signature: (v4i AS1* src, v4i AS3* dst, imm offset, imm cpol)
#define VQ_AS1_V4I(p) ((__attribute__((address_space(1))) v4i*)(p))
#define VQ_AS3_V4I(p) ((__attribute__((address_space(3))) v4i*)(p))

__device__ __forceinline__ void vq_wait_async0() {
#if defined(__has_builtin)
#if __has_builtin(__builtin_amdgcn_s_wait_asynccnt)
    __builtin_amdgcn_s_wait_asynccnt(0);
    return;
#else
    asm volatile("s_wait_asynccnt 0x0" ::: "memory");
    return;
#endif
#else
    asm volatile("s_wait_asynccnt 0x0" ::: "memory");
#endif
}

// ---------------- workspace layout (bytes) ----------------
// [0)           x_bf16   : VQ_NROWS*VQ_D*2      = 16777216
// [16777216)    emb_bf16 : VQ_K*VQ_D*2          =  8388608
// [25165824)    e2       : VQ_K*4               =    32768
// [25198592)    segval   : VQ_NROWS*VQ_SEGS*4   =   262144
// [25460736)    segidx   : VQ_NROWS*VQ_SEGS*4   =   262144
// [25722880)    partial  : VQ_NROWS*4           =    65536
#define WS_XB   0
#define WS_EB   16777216
#define WS_E2   25165824
#define WS_SVAL 25198592
#define WS_SIDX 25460736
#define WS_PART 25722880

// ============ kernel 1: x fp32 -> bf16 ============
__global__ void vq_cvt_x(const float* __restrict__ x, __bf16* __restrict__ xb, int n) {
    int i = (blockIdx.x * blockDim.x + threadIdx.x) * 4;
    if (i + 3 < n) {
        float4 v = *(const float4*)(x + i);
        xb[i + 0] = (__bf16)v.x;
        xb[i + 1] = (__bf16)v.y;
        xb[i + 2] = (__bf16)v.z;
        xb[i + 3] = (__bf16)v.w;
    }
}

// ============ kernel 2: emb fp32 -> bf16, plus e2[k] = ||e_k||^2 ============
__global__ void vq_cvt_emb(const float* __restrict__ e, __bf16* __restrict__ eb,
                           float* __restrict__ e2) {
    const int k = blockIdx.x;
    const float* row = e + (size_t)k * VQ_D;
    __bf16* orow = eb + (size_t)k * VQ_D;
    float s = 0.0f;
    for (int d = threadIdx.x; d < VQ_D; d += 128) {
        float v = row[d];
        s += v * v;
        orow[d] = (__bf16)v;
    }
    for (int off = 16; off > 0; off >>= 1) s += __shfl_down(s, off, 32);
    __shared__ float red[4];
    if ((threadIdx.x & 31) == 0) red[threadIdx.x >> 5] = s;
    __syncthreads();
    if (threadIdx.x == 0) e2[k] = red[0] + red[1] + red[2] + red[3];
}

// ============ kernel 3: fused bf16 WMMA distance GEMM + segment argmin ============
// Score(n,k) = e2[k] - 2 * dot(x_n, e_k)  (x^2 is row-constant; argmin unchanged)
// A = codebook tile (M=16 codes), B = x tile (N=16 rows), contraction over D.
// blockIdx.y selects a 2048-code segment; per-row segment winners go to ws.
__global__ __launch_bounds__(256) void vq_argmin(
    const __bf16* __restrict__ xb, const __bf16* __restrict__ eb,
    const float* __restrict__ e2, float* __restrict__ sval, int* __restrict__ sidx) {

    __shared__ __align__(16) __bf16 At[2][16 * VQ_D];   // 32 KB double-buffered A tile

    const int tid  = threadIdx.x;
    const int lane = tid & 31;
    const int wave = tid >> 5;
    const int rit  = lane & 15;   // row within this wave's 16-row x tile
    const int hi   = lane >> 4;   // which half of the 16-lane striping
    const int xrow = blockIdx.x * 128 + wave * 16 + rit;
    const int seg  = blockIdx.y;
    const int kbase = seg * VQ_KSEG;
    const __bf16* ebseg = eb + (size_t)kbase * VQ_D;

    // B operand (x) register-resident: lane holds 32B runs of its row per chunk.
    v16bf B[VQ_CHUNKS];
    {
        const __bf16* p = xb + (size_t)xrow * VQ_D + hi * 16;
        #pragma unroll
        for (int c = 0; c < VQ_CHUNKS; ++c)
            B[c] = *(const v16bf*)(p + c * 32);
    }

    float bestv = 3.4e38f;
    int   besti = kbase;

#if VQ_ASYNC
    // ---- async global->LDS staging: 256 thr x 4 x b128 = 16 KB tile ----
    #pragma unroll
    for (int p = 0; p < 4; ++p)
        __builtin_amdgcn_global_load_async_to_lds_b128(
            VQ_AS1_V4I((const char*)ebseg + p * 4096 + tid * 16),
            VQ_AS3_V4I((char*)&At[0][0] + p * 4096 + tid * 16), 0, 0);
    vq_wait_async0();
    __syncthreads();
#else
    uint4 pf[4];
    {
        const uint4* g = (const uint4*)ebseg;
        #pragma unroll
        for (int p = 0; p < 4; ++p) pf[p] = g[p * 512 + tid];
        #pragma unroll
        for (int p = 0; p < 4; ++p) ((uint4*)At[0])[p * 512 + tid] = pf[p];
    }
    __syncthreads();
#endif

    for (int mt = 0; mt < VQ_TILES; ++mt) {
        const int buf = mt & 1;

        // stage next code tile while the WMMAs run
        if (mt + 1 < VQ_TILES) {
#if VQ_ASYNC
            const char* gs = (const char*)(ebseg + (size_t)(mt + 1) * 16 * VQ_D);
            #pragma unroll
            for (int p = 0; p < 4; ++p)
                __builtin_amdgcn_global_load_async_to_lds_b128(
                    VQ_AS1_V4I(gs + p * 4096 + tid * 16),
                    VQ_AS3_V4I((char*)&At[buf ^ 1][0] + p * 4096 + tid * 16), 0, 0);
#else
            const uint4* g = (const uint4*)(ebseg + (size_t)(mt + 1) * 16 * VQ_D);
            #pragma unroll
            for (int p = 0; p < 4; ++p) pf[p] = g[p * 512 + tid];
#endif
        }

        // 16 chained bf16 WMMAs: full D=512 contraction for a 16x16 tile
        v8f acc = {};
        const __bf16* arow = &At[buf][rit * VQ_D + hi * 16];
        #pragma unroll
        for (int c = 0; c < VQ_CHUNKS; ++c) {
            union { v16bf v; v8bf h[2]; } a;
            a.h[0] = *(const v8bf*)(arow + c * 32);        // K lo half for this lane
            a.h[1] = *(const v8bf*)(arow + c * 32 + 8);    // K hi half for this lane
            acc = __builtin_amdgcn_wmma_f32_16x16x32_bf16(
                false, a.v, false, B[c], (short)0, acc, false, false);
        }

        // argmin epilogue: lane owns x-row `rit`, codes mbase..mbase+7
        const int mbase = kbase + mt * 16 + hi * 8;
        #pragma unroll
        for (int v = 0; v < 8; ++v) {
            float score = e2[mbase + v] - 2.0f * acc[v];
            if (score < bestv) { bestv = score; besti = mbase + v; }
        }

        if (mt + 1 < VQ_TILES) {
#if VQ_ASYNC
            vq_wait_async0();      // our async writes have landed in LDS
            __syncthreads();       // everyone's have
#else
            #pragma unroll
            for (int p = 0; p < 4; ++p) ((uint4*)At[buf ^ 1])[p * 512 + tid] = pf[p];
            __syncthreads();
#endif
        }
    }

    // lanes n and n+16 cover disjoint code halves of row n: merge (tie -> lower idx)
    float ov = __shfl_xor(bestv, 16, 32);
    int   oi = __shfl_xor(besti, 16, 32);
    if (ov < bestv || (ov == bestv && oi < besti)) { bestv = ov; besti = oi; }
    if (hi == 0) {
        sval[(size_t)xrow * VQ_SEGS + seg] = bestv;
        sidx[(size_t)xrow * VQ_SEGS + seg] = besti;
    }
}

// ============ kernel 4: merge segments, gather emb[idx] -> output, SSE partial ============
__global__ void vq_gather(const float* __restrict__ x, const float* __restrict__ emb,
                          const float* __restrict__ sval, const int* __restrict__ sidx,
                          float* __restrict__ outq, float* __restrict__ outidx,
                          float* __restrict__ partial) {
    const int row = blockIdx.x;
    const int t = threadIdx.x;       // 128 threads
    __shared__ int ksel;
    if (t == 0) {
        float bv = sval[(size_t)row * VQ_SEGS];
        int   bi = sidx[(size_t)row * VQ_SEGS];
        #pragma unroll
        for (int s2 = 1; s2 < VQ_SEGS; ++s2) {
            float v = sval[(size_t)row * VQ_SEGS + s2];
            int   i2 = sidx[(size_t)row * VQ_SEGS + s2];
            if (v < bv || (v == bv && i2 < bi)) { bv = v; bi = i2; }
        }
        ksel = bi;
    }
    __syncthreads();
    const int k = ksel;

    const float* q  = emb + (size_t)k * VQ_D;
    const float* xr = x + (size_t)row * VQ_D;
    float* o = outq + (size_t)row * VQ_D;
    float s = 0.0f;
    #pragma unroll
    for (int i = 0; i < 4; ++i) {
        int d = t + i * 128;
        float qv = q[d], xv = xr[d];
        o[d] = qv;                   // quantized_st forward value == quantized
        float df = qv - xv;
        s += df * df;
    }
    for (int off = 16; off > 0; off >>= 1) s += __shfl_down(s, off, 32);
    __shared__ float red[4];
    if ((t & 31) == 0) red[t >> 5] = s;
    __syncthreads();
    if (t == 0) {
        partial[row] = red[0] + red[1] + red[2] + red[3];
        outidx[row] = (float)k;
    }
}

// ============ kernel 5: deterministic loss reduction ============
__global__ void vq_loss(const float* __restrict__ partial, float* __restrict__ out) {
    float s = 0.0f;
    for (int i = threadIdx.x; i < VQ_NROWS; i += 256) s += partial[i];
    for (int off = 16; off > 0; off >>= 1) s += __shfl_down(s, off, 32);
    __shared__ float red[8];
    if ((threadIdx.x & 31) == 0) red[threadIdx.x >> 5] = s;
    __syncthreads();
    if (threadIdx.x == 0) {
        float tot = 0.0f;
        #pragma unroll
        for (int w = 0; w < 8; ++w) tot += red[w];
        float mse = tot / (float)(VQ_NROWS * VQ_D);
        out[0] = 0.25f * mse;   // commitment_loss
        out[1] = mse;           // codebook_loss
        out[2] = 1.25f * mse;   // total_loss
    }
}

extern "C" void kernel_launch(void* const* d_in, const int* in_sizes, int n_in,
                              void* d_out, int out_size, void* d_ws, size_t ws_size,
                              hipStream_t stream) {
    const float* x   = (const float*)d_in[0];   // [16,1024,512] f32
    const float* emb = (const float*)d_in[1];   // [8192,512]    f32

    char* ws = (char*)d_ws;
    __bf16* xb  = (__bf16*)(ws + WS_XB);
    __bf16* ebb = (__bf16*)(ws + WS_EB);
    float*  e2  = (float*)(ws + WS_E2);
    float*  sv  = (float*)(ws + WS_SVAL);
    int*    si  = (int*)(ws + WS_SIDX);
    float*  prt = (float*)(ws + WS_PART);

    float* out     = (float*)d_out;
    float* outq    = out;                               // [16,1024,512]
    float* outidx  = out + (size_t)VQ_NROWS * VQ_D;     // [16,1024] as float
    float* outloss = outidx + VQ_NROWS;                 // 3 scalars

    vq_cvt_x  <<<(VQ_NROWS * VQ_D) / 1024, 256, 0, stream>>>(x, xb, VQ_NROWS * VQ_D);
    vq_cvt_emb<<<VQ_K, 128, 0, stream>>>(emb, ebb, e2);
    vq_argmin <<<dim3(VQ_NROWS / 128, VQ_SEGS), 256, 0, stream>>>(xb, ebb, e2, sv, si);
    vq_gather <<<VQ_NROWS, 128, 0, stream>>>(x, emb, sv, si, outq, outidx, prt);
    vq_loss   <<<1, 256, 0, stream>>>(prt, outloss);
}
